// AttentionBase_40630390621000
// MI455X (gfx1250) — matrix operand order repaired and checked by reading, not compiled, not run on hardware
//
#include <hip/hip_runtime.h>
#include <hip/hip_bf16.h>
#include <cstdint>

typedef __attribute__((ext_vector_type(16))) _Float16 v16h;
typedef __attribute__((ext_vector_type(8)))  _Float16 v8h;
typedef __attribute__((ext_vector_type(4)))  _Float16 v4h;
typedef __attribute__((ext_vector_type(8)))  float    v8f;

#define B_   4
#define N_   2048
#define M_   2048
#define DIN_ 512
#define DOUT_ 512
#define H_   8
#define HD_  64
#define VD_  64
#define SCALE_ 0.125f           // 64^-0.5
#define KST  72                 // LDS row stride in halves (pad vs 64)
#define WST  520                // proj W LDS stride in halves (pad vs 512)

static __device__ inline v8f wmma16(v16h a, v16h b, v8f c) {
    return __builtin_amdgcn_wmma_f32_16x16x32_f16(false, a, false, b, (short)0, c, false, false);
}

// A-fragment (16x32 f16) from a global f32 row.
// row = lane&15, halves i -> K = d0 + 16*(lane>>4) + i (contiguous)
static __device__ inline v16h load_a_global_f32(const float* rowptr, int d0, int koff) {
    const float4* p0 = reinterpret_cast<const float4*>(rowptr + d0 + koff);
    const float4* p1 = reinterpret_cast<const float4*>(rowptr + d0 + 8 + koff);
    float4 x0 = p0[0], x1 = p0[1];
    float4 y0 = p1[0], y1 = p1[1];
    v16h a;
    a[0]=(_Float16)x0.x;  a[1]=(_Float16)x0.y;  a[2]=(_Float16)x0.z;  a[3]=(_Float16)x0.w;
    a[4]=(_Float16)x1.x;  a[5]=(_Float16)x1.y;  a[6]=(_Float16)x1.z;  a[7]=(_Float16)x1.w;
    a[8]=(_Float16)y0.x;  a[9]=(_Float16)y0.y;  a[10]=(_Float16)y0.z; a[11]=(_Float16)y0.w;
    a[12]=(_Float16)y1.x; a[13]=(_Float16)y1.y; a[14]=(_Float16)y1.z; a[15]=(_Float16)y1.w;
    return a;
}

// A-fragment from f16 row (LDS or global): two contiguous 8-half (b128) loads.
static __device__ inline v16h load_a_f16(const _Float16* rowptr, int d0, int koff) {
    v8h lo = *reinterpret_cast<const v8h*>(rowptr + d0 + koff);
    v8h hi = *reinterpret_cast<const v8h*>(rowptr + d0 + 8 + koff);
    v16h a;
#pragma unroll
    for (int i = 0; i < 8; ++i) { a[i] = lo[i]; a[8 + i] = hi[i]; }
    return a;
}

// B-fragment (32x16 f16): col = lane&15, halves i -> K = kstart + i (contiguous).
static __device__ inline v16h load_b_f16(const _Float16* colptr, int kstart) {
    v8h lo = *reinterpret_cast<const v8h*>(colptr + kstart);
    v8h hi = *reinterpret_cast<const v8h*>(colptr + kstart + 8);
    v16h b;
#pragma unroll
    for (int i = 0; i < 8; ++i) { b[i] = lo[i]; b[8 + i] = hi[i]; }
    return b;
}

// ---------------------------------------------------------------------------
// Pass 0: K -> f16 head-major [bh][M][64]; V -> f16 TRANSPOSED [bh][64][M]
// (LDS-transposed tile so global reads and writes both stay coalesced).
// grid = B*H*(M/64) = 1024, block = 256.
// ---------------------------------------------------------------------------
__global__ __launch_bounds__(256) void cvt_kv_kernel(
    const float* __restrict__ k, const float* __restrict__ v,
    _Float16* __restrict__ k16, _Float16* __restrict__ vt) {

    __shared__ _Float16 tl[64 * KST];

    const int tid = (int)threadIdx.x;
    const int j0 = (blockIdx.x & 31) * 64;
    const int bh = blockIdx.x >> 5;
    const int h = bh & 7;
    const int b = bh >> 3;
    const size_t kbase = ((size_t)b * M_) * DIN_ + h * HD_;
    const size_t vbase = ((size_t)b * M_) * DOUT_ + h * VD_;

#pragma unroll
    for (int i = 0; i < 4; ++i) {
        int idx = tid + i * 256;         // 1024 float4 per tile
        int m  = idx >> 4;
        int c4 = (idx & 15) * 4;
        float4 f = *reinterpret_cast<const float4*>(k + kbase + (size_t)(j0 + m) * DIN_ + c4);
        v4h hk; hk[0]=(_Float16)f.x; hk[1]=(_Float16)f.y; hk[2]=(_Float16)f.z; hk[3]=(_Float16)f.w;
        *reinterpret_cast<v4h*>(&k16[((size_t)bh * M_ + j0 + m) * 64 + c4]) = hk;
        float4 g = *reinterpret_cast<const float4*>(v + vbase + (size_t)(j0 + m) * DOUT_ + c4);
        tl[(c4 + 0) * KST + m] = (_Float16)g.x;
        tl[(c4 + 1) * KST + m] = (_Float16)g.y;
        tl[(c4 + 2) * KST + m] = (_Float16)g.z;
        tl[(c4 + 3) * KST + m] = (_Float16)g.w;
    }
    __syncthreads();
#pragma unroll
    for (int i = 0; i < 4; ++i) {
        int idx = tid + i * 256;
        int d  = idx >> 4;
        int m4 = (idx & 15) * 4;
        v4h t = *reinterpret_cast<v4h*>(&tl[d * KST + m4]);
        *reinterpret_cast<v4h*>(&vt[((size_t)bh * 64 + d) * M_ + j0 + m4]) = t;
    }
}

// ---------------------------------------------------------------------------
// Kernel 1: flash attention, double-buffered async K/V tiles.
// grid = B*H*(N/128), block = 256 (8 waves).  K tile [key][d], V tile [vd][key]
// copied raw (already f16) with global_load_async_to_lds_b128; the wave issues
// the next chunk's copy, runs 16 WMMAs + online softmax on the current one,
// then s_wait_asynccnt 0 + one barrier per chunk.
// ---------------------------------------------------------------------------
__global__ __launch_bounds__(256) void attn_fa_kernel(
    const float* __restrict__ q, const _Float16* __restrict__ k16,
    const _Float16* __restrict__ vt, _Float16* __restrict__ ows) {

    __shared__ _Float16 k_lds[2][64 * KST];
    __shared__ _Float16 v_lds[2][64 * KST];
    __shared__ _Float16 p_lds[8][16 * KST];

    const int lane = (int)threadIdx.x & 31;
    const int wave = (int)threadIdx.x >> 5;
    const int rt = blockIdx.x & 15;             // 16 row tiles of 128
    const int bh = blockIdx.x >> 4;
    const int h = bh & 7;
    const int b = bh >> 3;

    const int row0 = rt * 128 + wave * 16;
    const int col  = lane & 15;
    const int koff = (lane >> 4) * 16;
    const int k16o = (lane >> 4) * 16;

    // Q fragments (held in registers all kernel)
    const float* qrow = q + ((size_t)(b * N_ + row0 + col)) * DIN_ + h * HD_;
    v16h aq0 = load_a_global_f32(qrow, 0, koff);
    v16h aq1 = load_a_global_f32(qrow, 32, koff);

    float m_run[8], l_run[8];
    v8f acc_o[4];
#pragma unroll
    for (int r = 0; r < 8; ++r) { m_run[r] = -3.0e38f; l_run[r] = 0.0f; }
#pragma unroll
    for (int t = 0; t < 4; ++t) acc_o[t] = (v8f){};

    // async-copy lane assignment: 8 rows per wave, 4 segment-lanes per row
    const int crow = wave * 8 + (lane >> 2);    // tile row (key for K, vd for V)
    const int cseg = lane & 3;                  // 16B segment 0..3 (+4 via +64B)
    const _Float16* khead = k16 + (size_t)bh * M_ * 64;
    const _Float16* vhead = vt + (size_t)bh * 64 * M_;

    auto issue_copy = [&](int buf, int j0) {
        const _Float16* gk = khead + (size_t)(j0 + crow) * 64 + cseg * 8;
        uint32_t lk = (uint32_t)(uintptr_t)&k_lds[buf][crow * KST + cseg * 8];
        asm volatile("global_load_async_to_lds_b128 %0, %1, off"
                     :: "v"(lk), "v"(gk) : "memory");
        asm volatile("global_load_async_to_lds_b128 %0, %1, off"
                     :: "v"(lk + 64u), "v"(gk + 32) : "memory");
        const _Float16* gv = vhead + (size_t)crow * M_ + j0 + cseg * 8;
        uint32_t lv = (uint32_t)(uintptr_t)&v_lds[buf][crow * KST + cseg * 8];
        asm volatile("global_load_async_to_lds_b128 %0, %1, off"
                     :: "v"(lv), "v"(gv) : "memory");
        asm volatile("global_load_async_to_lds_b128 %0, %1, off"
                     :: "v"(lv + 64u), "v"(gv + 32) : "memory");
    };

    issue_copy(0, 0);
    asm volatile("s_wait_asynccnt 0" ::: "memory");
    __syncthreads();

    int cur = 0;
    for (int j0 = 0; j0 < M_; j0 += 64) {
        if (j0 + 64 < M_) issue_copy(cur ^ 1, j0 + 64);   // overlap next copy

        const _Float16* klc = &k_lds[cur][0];
        const _Float16* vlc = &v_lds[cur][0];

        // ---- S = Q * K^T : 4 key tiles x 2 K-slices = 8 WMMAs ----
        v8f s[4];
#pragma unroll
        for (int t = 0; t < 4; ++t) {
            s[t] = (v8f){};
            const _Float16* cb = klc + (t * 16 + col) * KST;
            v16h b0 = load_b_f16(cb, k16o);
            s[t] = wmma16(aq0, b0, s[t]);
            v16h b1 = load_b_f16(cb, 32 + k16o);
            s[t] = wmma16(aq1, b1, s[t]);
        }

        // ---- online softmax ----
        float mnew[8];
#pragma unroll
        for (int r = 0; r < 8; ++r) {
            float mm = m_run[r];
#pragma unroll
            for (int t = 0; t < 4; ++t) mm = fmaxf(mm, s[t][r] * SCALE_);
            mm = fmaxf(mm, __shfl_xor(mm, 1, 32));
            mm = fmaxf(mm, __shfl_xor(mm, 2, 32));
            mm = fmaxf(mm, __shfl_xor(mm, 4, 32));
            mm = fmaxf(mm, __shfl_xor(mm, 8, 32));
            mnew[r] = mm;
        }

        _Float16* pw = &p_lds[wave][0];
#pragma unroll
        for (int r = 0; r < 8; ++r) {
            float corr = __expf(m_run[r] - mnew[r]);
            m_run[r] = mnew[r];
            l_run[r] *= corr;
#pragma unroll
            for (int t = 0; t < 4; ++t) acc_o[t][r] *= corr;
            float rs = 0.0f;
            int rowl = r + (lane >> 4) * 8;
#pragma unroll
            for (int t = 0; t < 4; ++t) {
                float pv = __expf(s[t][r] * SCALE_ - mnew[r]);
                rs += pv;
                pw[rowl * KST + t * 16 + col] = (_Float16)pv;
            }
            rs += __shfl_xor(rs, 1, 32);
            rs += __shfl_xor(rs, 2, 32);
            rs += __shfl_xor(rs, 4, 32);
            rs += __shfl_xor(rs, 8, 32);
            l_run[r] += rs;
        }

        // ---- O += P * V : 2 key slices x 4 vd tiles = 8 WMMAs ----
        const _Float16* prow = &p_lds[wave][col * KST];
#pragma unroll
        for (int s2 = 0; s2 < 2; ++s2) {
            v16h ap = load_a_f16(prow, s2 * 32, koff);
#pragma unroll
            for (int vtl = 0; vtl < 4; ++vtl) {
                const _Float16* vb = vlc + (vtl * 16 + col) * KST;
                v16h bv = load_b_f16(vb, s2 * 32 + k16o);
                acc_o[vtl] = wmma16(ap, bv, acc_o[vtl]);
            }
        }

        asm volatile("s_wait_asynccnt 0" ::: "memory");  // next buffers landed
        __syncthreads();                                 // all waves done w/ cur
        cur ^= 1;
    }

    // ---- normalize and store f16 to workspace: [b*N+n][h*64+vd] ----
#pragma unroll
    for (int r = 0; r < 8; ++r) {
        float inv = 1.0f / l_run[r];
        int rowl = r + (lane >> 4) * 8;
        size_t ob = ((size_t)(b * N_ + row0 + rowl)) * DOUT_ + h * VD_;
#pragma unroll
        for (int vtl = 0; vtl < 4; ++vtl)
            ows[ob + vtl * 16 + col] = (_Float16)(acc_o[vtl][r] * inv);
    }
}

// ---------------------------------------------------------------------------
// Kernel 2: out = A @ W^T + b.  A: [8192,512] f16 (ws), W: [512,512] f32.
// grid = (8192/128)*(512/32) = 1024, block = 256 (8 waves), 32 WMMAs/wave.
// ---------------------------------------------------------------------------
__global__ __launch_bounds__(256) void proj_kernel(
    const _Float16* __restrict__ aws, const float* __restrict__ W,
    const float* __restrict__ bias, float* __restrict__ out) {

    __shared__ _Float16 w_lds[32 * WST];   // [o_local][d]

    const int lane = (int)threadIdx.x & 31;
    const int wave = (int)threadIdx.x >> 5;
    const int cg = blockIdx.x & 15;        // 16 col groups of 32
    const int rt = blockIdx.x >> 4;        // 64 row tiles of 128
    const int obase = cg * 32;
    const int col  = lane & 15;
    const int koff = (lane >> 4) * 16;
    const int k16o = (lane >> 4) * 16;

#pragma unroll
    for (int i = 0; i < 16; ++i) {
        int idx = (int)threadIdx.x + i * 256;   // 4096 float4 total
        int ol = idx >> 7;
        int c4 = (idx & 127) * 4;
        float4 f = *reinterpret_cast<const float4*>(W + (size_t)(obase + ol) * DOUT_ + c4);
        v4h hw; hw[0]=(_Float16)f.x; hw[1]=(_Float16)f.y; hw[2]=(_Float16)f.z; hw[3]=(_Float16)f.w;
        *reinterpret_cast<v4h*>(&w_lds[ol * WST + c4]) = hw;
    }
    __syncthreads();

    const int row = rt * 128 + wave * 16 + col;
    const _Float16* arow = aws + (size_t)row * DOUT_;

    v8f acc0 = (v8f){}, acc1 = (v8f){};
#pragma unroll
    for (int d0 = 0; d0 < DOUT_; d0 += 32) {
        v16h a  = load_a_f16(arow, d0, koff);
        v16h b0 = load_b_f16(&w_lds[(col) * WST], d0 + k16o);
        acc0 = wmma16(a, b0, acc0);
        v16h b1 = load_b_f16(&w_lds[(16 + col) * WST], d0 + k16o);
        acc1 = wmma16(a, b1, acc1);
    }

    float bb0 = bias[obase + col];
    float bb1 = bias[obase + 16 + col];
#pragma unroll
    for (int r = 0; r < 8; ++r) {
        int orow = rt * 128 + wave * 16 + r + (lane >> 4) * 8;
        out[(size_t)orow * DOUT_ + obase + col]      = acc0[r] + bb0;
        out[(size_t)orow * DOUT_ + obase + 16 + col] = acc1[r] + bb1;
    }
}

extern "C" void kernel_launch(void* const* d_in, const int* in_sizes, int n_in,
                              void* d_out, int out_size, void* d_ws, size_t ws_size,
                              hipStream_t stream) {
    const float* q    = (const float*)d_in[0];
    const float* k    = (const float*)d_in[1];
    const float* v    = (const float*)d_in[2];
    const float* W    = (const float*)d_in[3];
    const float* bias = (const float*)d_in[4];
    float* out = (float*)d_out;

    // workspace map (24 MB total): k16 | vT | attention output
    const size_t S1 = (size_t)B_ * H_ * M_ * 64 * sizeof(_Float16);   // 8.4 MB each
    _Float16* k16 = (_Float16*)d_ws;
    _Float16* vt  = (_Float16*)((char*)d_ws + S1);
    _Float16* aws = (_Float16*)((char*)d_ws + 2 * S1);

    cvt_kv_kernel<<<dim3(B_ * H_ * (M_ / 64)), dim3(256), 0, stream>>>(k, v, k16, vt);
    attn_fa_kernel<<<dim3(B_ * H_ * (N_ / 128)), dim3(256), 0, stream>>>(q, k16, vt, aws);
    proj_kernel<<<dim3((B_ * N_ / 128) * (DOUT_ / 32)), dim3(256), 0, stream>>>(aws, W, bias, out);
}
